// LSTM_28432683499547
// MI455X (gfx1250) — 16-bit and 8-bit matrix operand order repaired and checked by reading, compiled, not run on hardware
//
#include <hip/hip_runtime.h>

#define T_LEN 2048
#define D_IN  512
#define HID   512
#define HD    256
#define G4    1024
#define NTAGS 20
#define START_TAG 18
#define END_TAG   19
#define NEGV  -10000.0f

typedef __attribute__((ext_vector_type(16))) __bf16 v16bf;
typedef __attribute__((ext_vector_type(8)))  float  v8f;
typedef __attribute__((ext_vector_type(8)))  int    v8i;

__device__ inline unsigned int fp8_of(float x) {
#if __has_builtin(__builtin_amdgcn_cvt_pk_fp8_f32)
  return ((unsigned)__builtin_amdgcn_cvt_pk_fp8_f32(x, 0.0f, 0, false)) & 0xFFu;
#else
  unsigned u = __float_as_uint(x);
  unsigned s = (u >> 24) & 0x80u;
  unsigned ax = u & 0x7FFFFFFFu;
  if (ax == 0u || ax >= 0x7F800000u) return s;
  int e = (int)(ax >> 23) - 127;
  unsigned m = (ax >> 20) & 0x7u;
  if (e > 8 || (e == 8 && m > 6)) return s | 0x7Eu;
  if (e < -9) return s;
  if (e < -6) { unsigned mm = (0x8u | m) >> (unsigned)(-6 - e); return s | mm; }
  return s | ((unsigned)(e + 7) << 3) | m;
#endif
}

__global__ void embed_gather(const int* __restrict__ sent,
                             const float* __restrict__ emb,
                             float* __restrict__ x0) {
  int t = blockIdx.x;
  const float* src = emb + (size_t)sent[t] * D_IN;
  float* dst = x0 + (size_t)t * D_IN;
  for (int d = threadIdx.x; d < D_IN; d += blockDim.x) dst[d] = src[d];
}

__global__ void pack_whh_fp8(const float* __restrict__ W0, const float* __restrict__ W1,
                             const float* __restrict__ W2, const float* __restrict__ W3,
                             unsigned char* __restrict__ out) {
  const float* W = (blockIdx.y == 0) ? W0 : (blockIdx.y == 1) ? W1
                  : (blockIdx.y == 2) ? W2 : W3;
  unsigned char* dst = out + (size_t)blockIdx.y * (64u * 4u * 32u * 32u);
  unsigned idx = blockIdx.x * 256u + threadIdx.x;
  unsigned b    =  idx        & 3u;
  unsigned v    = (idx >> 2)  & 7u;
  unsigned lane = (idx >> 5)  & 31u;
  unsigned kk   = (idx >> 10) & 3u;
  unsigned tile =  idx >> 12;
  unsigned g = lane >> 4;
  unsigned m = tile * 16u + (lane & 15u);
  unsigned k = kk * 64u + 16u * g + ((v & 1u) ? 4u : 0u) + ((v & 2u) ? 8u : 0u)
             + ((v & 4u) ? 32u : 0u) + b;
  dst[idx] = (unsigned char)fp8_of(W[(size_t)m * HD + k]);
}

__global__ void __launch_bounds__(32)
gemm_bf16_wmma(const float* __restrict__ X, const float* __restrict__ W,
               const float* __restrict__ bias, float* __restrict__ Y,
               int K, int N) {
  int bx = blockIdx.x, tm = blockIdx.y;
  int lane = threadIdx.x;
  int g = lane >> 4, l = lane & 15;
  int tn0 = 2 * bx, tn1 = 2 * bx + 1;
  const float* xrow = X + (size_t)(tm * 16 + l) * K;
  const float* w0   = W + (size_t)(tn0 * 16 + l) * K;
  const float* w1   = W + (size_t)(tn1 * 16 + l) * K;
  v8f acc0 = {}, acc1 = {};
  for (int k0 = 0; k0 < K; k0 += 32) {
    v16bf a, b0, b1;
    int ka = k0 + 16 * g;
#pragma unroll
    for (int e = 0; e < 8; ++e) {
      a[e]     = (__bf16)xrow[ka + e];
      a[e + 8] = (__bf16)xrow[ka + 8 + e];
    }
    int kb = k0 + 16 * g;
#pragma unroll
    for (int e = 0; e < 16; ++e) { b0[e] = (__bf16)w0[kb + e]; b1[e] = (__bf16)w1[kb + e]; }
    acc0 = __builtin_amdgcn_wmma_f32_16x16x32_bf16(false, a, false, b0,
                                                   (short)0, acc0, false, false);
    acc1 = __builtin_amdgcn_wmma_f32_16x16x32_bf16(false, a, false, b1,
                                                   (short)0, acc1, false, false);
  }
  float bn0 = bias[tn0 * 16 + l];
  float bn1 = bias[tn1 * 16 + l];
#pragma unroll
  for (int r = 0; r < 8; ++r) {
    int row = tm * 16 + 8 * g + r;
    Y[(size_t)row * N + tn0 * 16 + l] = acc0[r] + bn0;
    Y[(size_t)row * N + tn1 * 16 + l] = acc1[r] + bn1;
  }
}

__global__ void __launch_bounds__(1024)
lstm_dir(const float* __restrict__ xgF, const float* __restrict__ xgB,
         const unsigned char* __restrict__ wfF, const unsigned char* __restrict__ wfB,
         const float* __restrict__ bhhF, const float* __restrict__ bhhB,
         float* __restrict__ out) {
  __shared__ float zsh[G4];
  __shared__ float xbuf[2][G4];
  __shared__ __align__(16) unsigned int h8[HD / 4];

  int dir = blockIdx.x;
  const float* xg  = dir ? xgB : xgF;
  const unsigned char* wf = dir ? wfB : wfF;
  const float* bhh = dir ? bhhB : bhhF;
  int colOff = dir ? HD : 0;

  int tid = threadIdx.x, wave = tid >> 5, lane = tid & 31, g = lane >> 4;
  unsigned long long xgBase = (unsigned long long)(uintptr_t)xg;

  v8i afrag[2][4];
#pragma unroll
  for (int tt = 0; tt < 2; ++tt)
#pragma unroll
    for (int kk = 0; kk < 4; ++kk)
      afrag[tt][kk] = *(const v8i*)(wf + ((size_t)(((2 * wave + tt) * 4 + kk) * 32 + lane) * 32));

  float c = 0.f, b0 = 0.f, b1 = 0.f, b2 = 0.f, b3 = 0.f;
  if (tid < HD) { b0 = bhh[tid]; b1 = bhh[tid + 256]; b2 = bhh[tid + 512]; b3 = bhh[tid + 768]; }
  if (tid < HD / 4) h8[tid] = 0u;

  {
    int t0 = dir ? (T_LEN - 1) : 0;
    unsigned lds0 = (unsigned)(uintptr_t)&xbuf[0][tid];
    unsigned off0 = ((unsigned)t0 * G4 + (unsigned)tid) * 4u;
    asm volatile("global_load_async_to_lds_b32 %0, %1, %2"
                 :: "v"(lds0), "v"(off0), "s"(xgBase) : "memory");
    asm volatile("s_wait_asynccnt 0x0" ::: "memory");
  }
  __syncthreads();

  for (int step = 0; step < T_LEN; ++step) {
    int t  = dir ? (T_LEN - 1 - step) : step;
    int cb = step & 1, nb = cb ^ 1;

    {
      int tn = t + (dir ? -1 : 1);
      tn = (tn < 0) ? 0 : (tn >= T_LEN ? T_LEN - 1 : tn);
      unsigned ldsn = (unsigned)(uintptr_t)&xbuf[nb][tid];
      unsigned offn = ((unsigned)tn * G4 + (unsigned)tid) * 4u;
      asm volatile("global_load_async_to_lds_b32 %0, %1, %2"
                   :: "v"(ldsn), "v"(offn), "s"(xgBase) : "memory");
    }

    v8f acc0 = {}, acc1 = {};
    const unsigned char* hb = (const unsigned char*)h8;
#pragma unroll
    for (int kk = 0; kk < 4; ++kk) {
      int base = kk * 64 + 16 * g;
      uint4 blo = *(const uint4*)(hb + base);
      uint4 bhi = *(const uint4*)(hb + base + 32);
      v8i bf;
      bf[0] = (int)blo.x; bf[1] = (int)blo.y; bf[2] = (int)blo.z; bf[3] = (int)blo.w;
      bf[4] = (int)bhi.x; bf[5] = (int)bhi.y; bf[6] = (int)bhi.z; bf[7] = (int)bhi.w;
      acc0 = __builtin_amdgcn_wmma_f32_16x16x64_fp8_fp8(afrag[0][kk], bf, (short)0, acc0, false, false);
      acc1 = __builtin_amdgcn_wmma_f32_16x16x64_fp8_fp8(afrag[1][kk], bf, (short)0, acc1, false, false);
    }
    if ((lane & 15) == 0) {
      int r0 = 2 * wave * 16 + 8 * g, r1 = (2 * wave + 1) * 16 + 8 * g;
#pragma unroll
      for (int r = 0; r < 8; ++r) { zsh[r0 + r] = acc0[r]; zsh[r1 + r] = acc1[r]; }
    }
    __syncthreads();

    if (tid < HD) {
      const float* xt = xbuf[cb];
      float zi = xt[tid]       + b0 + zsh[tid];
      float zf = xt[tid + 256] + b1 + zsh[tid + 256];
      float zg = xt[tid + 512] + b2 + zsh[tid + 512];
      float zo = xt[tid + 768] + b3 + zsh[tid + 768];
      float ig = 1.f / (1.f + __expf(-zi));
      float fg = 1.f / (1.f + __expf(-zf));
      float og = 1.f / (1.f + __expf(-zo));
      c = fg * c + ig * tanhf(zg);
      float h = og * tanhf(c);
      out[(size_t)t * HID + colOff + tid] = h;
      ((unsigned char*)h8)[tid] = (unsigned char)fp8_of(h);
    }
    asm volatile("s_wait_asynccnt 0x0" ::: "memory");
    __syncthreads();
  }
}

__global__ void out_proj(const float* __restrict__ X, const float* __restrict__ W,
                         const float* __restrict__ b, float* __restrict__ F) {
  int t = blockIdx.x, n = threadIdx.x;
  if (n >= NTAGS) return;
  const float* x = X + (size_t)t * HID;
  const float* w = W + (size_t)n * HID;
  float s = b[n];
  for (int k = 0; k < HID; ++k) s += x[k] * w[k];
  F[t * NTAGS + n] = s;
}

__global__ void viterbi(const float* __restrict__ F, const float* __restrict__ Tr,
                        int* __restrict__ bp, float* __restrict__ outp) {
  __shared__ float v[NTAGS], vn[NTAGS];
  __shared__ float tr[NTAGS * NTAGS];
  int tid = threadIdx.x;
  for (int i = tid; i < NTAGS * NTAGS; i += 32) tr[i] = Tr[i];
  if (tid < NTAGS) v[tid] = (tid == START_TAG) ? 0.f : NEGV;
  __syncthreads();
  for (int t = 0; t < T_LEN; ++t) {
    if (tid < NTAGS) {
      float best = -3.4e38f; int bi = 0;
      for (int i = 0; i < NTAGS; ++i) {
        float s = v[i] + tr[tid * NTAGS + i];
        if (s > best) { best = s; bi = i; }
      }
      bp[t * NTAGS + tid] = bi;
      vn[tid] = best + F[t * NTAGS + tid];
    }
    __syncthreads();
    if (tid < NTAGS) v[tid] = vn[tid];
    __syncthreads();
  }
  if (tid == 0) {
    float best = -3.4e38f; int bt = 0;
    for (int j = 0; j < NTAGS; ++j) {
      float s = v[j] + tr[END_TAG * NTAGS + j];
      if (s > best) { best = s; bt = j; }
    }
    outp[0] = best;
    int tag = bt;
    for (int t = T_LEN - 1; t >= 0; --t) {
      outp[1 + t] = (float)tag;
      tag = bp[t * NTAGS + tag];
    }
  }
}

extern "C" void kernel_launch(void* const* d_in, const int* in_sizes, int n_in,
                              void* d_out, int out_size, void* d_ws, size_t ws_size,
                              hipStream_t stream) {
  const int*   sentence = (const int*)d_in[0];
  const float* emb      = (const float*)d_in[1];
  const float* Wih[4] = { (const float*)d_in[2],  (const float*)d_in[6],
                          (const float*)d_in[10], (const float*)d_in[14] };
  const float* Whh[4] = { (const float*)d_in[3],  (const float*)d_in[7],
                          (const float*)d_in[11], (const float*)d_in[15] };
  const float* bih[4] = { (const float*)d_in[4],  (const float*)d_in[8],
                          (const float*)d_in[12], (const float*)d_in[16] };
  const float* bhh[4] = { (const float*)d_in[5],  (const float*)d_in[9],
                          (const float*)d_in[13], (const float*)d_in[17] };
  const float* W_out = (const float*)d_in[18];
  const float* b_out = (const float*)d_in[19];
  const float* trans = (const float*)d_in[20];
  float* out = (float*)d_out;

  char* ws = (char*)d_ws;
  float* x0  = (float*)ws;              ws += (size_t)T_LEN * HID * 4;
  float* x1  = (float*)ws;              ws += (size_t)T_LEN * HID * 4;
  float* xgf = (float*)ws;              ws += (size_t)T_LEN * G4 * 4;
  float* xgb = (float*)ws;              ws += (size_t)T_LEN * G4 * 4;
  unsigned char* wfrag = (unsigned char*)ws; ws += (size_t)4 * 64 * 4 * 32 * 32;
  float* feats = (float*)ws;            ws += (size_t)T_LEN * NTAGS * 4;
  int*   bp    = (int*)ws;              ws += (size_t)T_LEN * NTAGS * 4;

  const size_t FRAG = (size_t)64 * 4 * 32 * 32;

  embed_gather<<<T_LEN, 256, 0, stream>>>(sentence, emb, x0);
  pack_whh_fp8<<<dim3(1024, 4), 256, 0, stream>>>(Whh[0], Whh[1], Whh[2], Whh[3], wfrag);

  dim3 gemmGrid(G4 / 32, T_LEN / 16);
  gemm_bf16_wmma<<<gemmGrid, 32, 0, stream>>>(x0, Wih[0], bih[0], xgf, D_IN, G4);
  gemm_bf16_wmma<<<gemmGrid, 32, 0, stream>>>(x0, Wih[1], bih[1], xgb, D_IN, G4);
  lstm_dir<<<2, 1024, 0, stream>>>(xgf, xgb, wfrag + 0 * FRAG, wfrag + 1 * FRAG,
                                   bhh[0], bhh[1], x1);
  gemm_bf16_wmma<<<gemmGrid, 32, 0, stream>>>(x1, Wih[2], bih[2], xgf, HID, G4);
  gemm_bf16_wmma<<<gemmGrid, 32, 0, stream>>>(x1, Wih[3], bih[3], xgb, HID, G4);
  lstm_dir<<<2, 1024, 0, stream>>>(xgf, xgb, wfrag + 2 * FRAG, wfrag + 3 * FRAG,
                                   bhh[2], bhh[3], x0);

  out_proj<<<T_LEN, 32, 0, stream>>>(x0, W_out, b_out, feats);
  viterbi<<<1, 32, 0, stream>>>(feats, trans, bp, out);
}
